// MyNet_3075196584646
// MI455X (gfx1250) — matrix operand order repaired and checked by reading, not compiled, not run on hardware
//
#include <hip/hip_runtime.h>

// ---------------- problem constants (from reference) ----------------
#define NN    12288      // nodes
#define EE    196608     // edges
#define FIN   1433       // input features
#define KPAD  1440       // FIN padded to multiple of 32
#define NCH   45         // KPAD / 32 K-chunks
#define H1    32         // GCN hidden
#define H2    16         // SAGE hidden
#define NC    7          // classes

typedef __attribute__((ext_vector_type(16))) __bf16 v16bf;
typedef __attribute__((ext_vector_type(8)))  float  v8f;

union Frag16 { uint4 q[2]; v16bf v; };

// float -> bf16 round-to-nearest-even
__device__ __forceinline__ unsigned short f2bf(float f) {
    unsigned int u = __float_as_uint(f);
    unsigned int r = u + 0x7FFFu + ((u >> 16) & 1u);
    return (unsigned short)(r >> 16);
}

// ---------------- workspace byte offsets ----------------
#define O_XBF   0u                       // [NN][KPAD] bf16 (ushort)       35,389,440 B
#define O_FRAG  35389440u                // packed W1 B-fragments              92,160 B
#define O_XW    35481600u                // xw = x@W1   [NN][32] f32        1,572,864 B
#define O_H     37054464u                // h           [NN][32] f32        1,572,864 B
#define O_AGG   38627328u                // agg_sum     [NN][32] f32        1,572,864 B
#define O_DEG   40200192u                // deg [NN] f32
#define O_CNT   40249344u                // cnt [NN] f32
#define O_DIS   40298496u                // dis [NN] f32

// =====================================================================
// 1) convert x -> bf16, K-padded to 1440 (zero pad)
// =====================================================================
__global__ void k_conv_x(const float* __restrict__ x, unsigned short* __restrict__ xbf) {
    unsigned int t = blockIdx.x * blockDim.x + threadIdx.x;   // NN*KPAD threads
    unsigned int m = t / KPAD, k = t % KPAD;
    float v = (k < FIN) ? x[(size_t)m * FIN + k] : 0.0f;
    xbf[t] = f2bf(v);
}

// =====================================================================
// 2) pre-pack W1 into exact WMMA B-fragment layout (bf16, zero K-pad).
//    Frag layout: [chunk][2 N-halves][lane 0..31][16 bf16]
//    B 32x16 bf16, contiguous k order used here (A matches): lane = kh*16 + n,  element i -> K = kh*16 + i, N = n
// =====================================================================
__global__ void k_pack_w1(const float* __restrict__ W1, unsigned short* __restrict__ frag) {
    unsigned int t = blockIdx.x * blockDim.x + threadIdx.x;   // NCH*2*32*16 threads
    unsigned int i    = t & 15;
    unsigned int lane = (t >> 4) & 31;
    unsigned int nf   = (t >> 9) & 1;
    unsigned int c    = t >> 10;
    unsigned int kh   = lane >> 4;
    unsigned int n    = nf * 16 + (lane & 15);
    unsigned int k    = c * 32 + kh * 16 + i;
    float v = (k < FIN) ? W1[k * H1 + n] : 0.0f;
    frag[t] = f2bf(v);
}

// =====================================================================
// 3) init: deg=1 (self loop), cnt=0, agg=0
// =====================================================================
__global__ void k_init(float* __restrict__ deg, float* __restrict__ cnt,
                       float* __restrict__ agg) {
    unsigned int t = blockIdx.x * blockDim.x + threadIdx.x;   // NN*32 threads
    agg[t] = 0.0f;
    if ((t & 31) == 0) { deg[t >> 5] = 1.0f; cnt[t >> 5] = 0.0f; }
}

// 4) degree accumulate over col indices
__global__ void k_deg(const int* __restrict__ ei, float* __restrict__ deg) {
    unsigned int e = blockIdx.x * blockDim.x + threadIdx.x;   // EE threads
    atomicAdd(&deg[ei[EE + e]], 1.0f);
}

// 5) dis = deg^-1/2
__global__ void k_dis(const float* __restrict__ deg, float* __restrict__ dis) {
    unsigned int i = blockIdx.x * blockDim.x + threadIdx.x;   // NN threads
    dis[i] = rsqrtf(deg[i]);
}

// =====================================================================
// 6) xw = x @ W1 via v_wmma_f32_16x16x32_bf16.
//    Block = 128 threads (4 waves); each wave owns a 16x32 output tile.
//    B fragments staged in LDS (92160 B), A streamed as 2x b128/lane/chunk.
// =====================================================================
__global__ void k_gemm(const unsigned short* __restrict__ xbf,
                       const unsigned short* __restrict__ w1frag,
                       float* __restrict__ xw) {
    extern __shared__ uint4 smemq[];                      // 92160 B = 5760 uint4
    // cooperative stage of all B fragments
    const uint4* src = (const uint4*)w1frag;
    for (unsigned int i = threadIdx.x; i < 5760; i += 128) smemq[i] = src[i];
    __syncthreads();

    const unsigned int wave = threadIdx.x >> 5;
    const unsigned int lane = threadIdx.x & 31;
    const unsigned int ml   = lane & 15;
    const unsigned int kh   = lane >> 4;
    const unsigned int tileM = blockIdx.x * 64 + wave * 16;

    // A 16x32 bf16, contiguous k order used here (matches packed B): lane kh half -> K = {kh*16..kh*16+15}
    const unsigned short* arow = xbf + (size_t)(tileM + ml) * KPAD + kh * 16;
    const unsigned char*  sb   = (const unsigned char*)smemq;

    v8f acc0 = {}; v8f acc1 = {};
    for (int c = 0; c < NCH; ++c) {
        Frag16 A, B0, B1;
        const unsigned short* ap = arow + c * 32;
        A.q[0] = *(const uint4*)(ap);
        A.q[1] = *(const uint4*)(ap + 8);
        const unsigned char* bp = sb + c * 2048 + lane * 32;
        B0.q[0] = ((const uint4*)bp)[0];
        B0.q[1] = ((const uint4*)bp)[1];
        B1.q[0] = ((const uint4*)(bp + 1024))[0];
        B1.q[1] = ((const uint4*)(bp + 1024))[1];
        acc0 = __builtin_amdgcn_wmma_f32_16x16x32_bf16(false, A.v, false, B0.v,
                                                       (short)0, acc0, false, false);
        acc1 = __builtin_amdgcn_wmma_f32_16x16x32_bf16(false, A.v, false, B1.v,
                                                       (short)0, acc1, false, false);
    }
    // C/D layout: lane<16 -> M = vgpr, N = lane ; lane>=16 -> M = vgpr+8
    for (int v = 0; v < 8; ++v) {
        unsigned int m = tileM + v + kh * 8;
        xw[(size_t)m * H1 + ml]      = acc0[v];
        xw[(size_t)m * H1 + 16 + ml] = acc1[v];
    }
}

// 7) self-loop term: h[i] = dis[i]^2 * xw[i]
__global__ void k_selfh(const float* __restrict__ dis, const float* __restrict__ xw,
                        float* __restrict__ h) {
    unsigned int t = blockIdx.x * blockDim.x + threadIdx.x;   // NN*32 threads
    float d = dis[t >> 5];
    h[t] = d * d * xw[t];
}

// 8) GCN edge scatter: h[r] += dis[r]*dis[c]*xw[c]
__global__ void k_gcn_edges(const int* __restrict__ ei, const float* __restrict__ dis,
                            const float* __restrict__ xw, float* __restrict__ h) {
    unsigned int t = blockIdx.x * blockDim.x + threadIdx.x;   // EE*8 threads
    unsigned int e = t >> 3, q = (t & 7) * 4;
    int r = ei[e], c = ei[EE + e];
    float w = dis[r] * dis[c];
    const float4 v = *(const float4*)(xw + (size_t)c * H1 + q);
    float* hp = h + (size_t)r * H1 + q;
    atomicAdd(hp + 0, w * v.x);
    atomicAdd(hp + 1, w * v.y);
    atomicAdd(hp + 2, w * v.z);
    atomicAdd(hp + 3, w * v.w);
}

// 9) h = relu(h + b1)
__global__ void k_bias_relu(float* __restrict__ h, const float* __restrict__ b1) {
    unsigned int t = blockIdx.x * blockDim.x + threadIdx.x;   // NN*32 threads
    h[t] = fmaxf(h[t] + b1[t & 31], 0.0f);
}

// 10) SAGE edge scatter: agg[src] += h[dst]; cnt[src] += 1
__global__ void k_sage_edges(const int* __restrict__ ei, const float* __restrict__ h,
                             float* __restrict__ agg, float* __restrict__ cnt) {
    unsigned int t = blockIdx.x * blockDim.x + threadIdx.x;   // EE*8 threads
    unsigned int e = t >> 3, q = (t & 7) * 4;
    int s = ei[e], d = ei[EE + e];
    const float4 v = *(const float4*)(h + (size_t)d * H1 + q);
    float* ap = agg + (size_t)s * H1 + q;
    atomicAdd(ap + 0, v.x);
    atomicAdd(ap + 1, v.y);
    atomicAdd(ap + 2, v.z);
    atomicAdd(ap + 3, v.w);
    if ((t & 7) == 0) atomicAdd(&cnt[s], 1.0f);
}

// =====================================================================
// 11) per-node tail: out = relu(h@Wl+bl + mean@Wr+br); L2-norm; W3 head; softmax
// =====================================================================
__global__ void k_final(const float* __restrict__ h, const float* __restrict__ agg,
                        const float* __restrict__ cnt,
                        const float* __restrict__ Wl, const float* __restrict__ bl,
                        const float* __restrict__ Wr, const float* __restrict__ br,
                        const float* __restrict__ W3, const float* __restrict__ b3,
                        float* __restrict__ out) {
    __shared__ float sWl[H1 * H2], sWr[H1 * H2], sW3[H2 * NC];
    __shared__ float sbl[H2], sbr[H2], sb3[NC];
    for (unsigned int i = threadIdx.x; i < H1 * H2; i += blockDim.x) { sWl[i] = Wl[i]; sWr[i] = Wr[i]; }
    for (unsigned int i = threadIdx.x; i < H2 * NC; i += blockDim.x) sW3[i] = W3[i];
    if (threadIdx.x < H2) { sbl[threadIdx.x] = bl[threadIdx.x]; sbr[threadIdx.x] = br[threadIdx.x]; }
    if (threadIdx.x < NC) sb3[threadIdx.x] = b3[threadIdx.x];
    __syncthreads();

    unsigned int i = blockIdx.x * blockDim.x + threadIdx.x;   // NN threads
    float hv[H1], av[H1];
    const float4* hp = (const float4*)(h   + (size_t)i * H1);
    const float4* ap = (const float4*)(agg + (size_t)i * H1);
    float cn = cnt[i];
    float sc = (cn > 0.0f) ? (1.0f / cn) : 0.0f;
    for (int k = 0; k < 8; ++k) {
        float4 a = hp[k]; float4 b = ap[k];
        hv[4*k+0] = a.x; hv[4*k+1] = a.y; hv[4*k+2] = a.z; hv[4*k+3] = a.w;
        av[4*k+0] = b.x * sc; av[4*k+1] = b.y * sc; av[4*k+2] = b.z * sc; av[4*k+3] = b.w * sc;
    }
    float o[H2];
    float nrm = 0.0f;
    for (int j = 0; j < H2; ++j) {
        float acc = sbl[j] + sbr[j];
        for (int k = 0; k < H1; ++k)
            acc = fmaf(hv[k], sWl[k * H2 + j], fmaf(av[k], sWr[k * H2 + j], acc));
        acc = fmaxf(acc, 0.0f);
        o[j] = acc;
        nrm += acc * acc;
    }
    float inv = 1.0f / (sqrtf(nrm) + 1e-6f);
    float lg[NC], mx = -3.0e38f;
    for (int j = 0; j < NC; ++j) {
        float acc = sb3[j];
        for (int k = 0; k < H2; ++k) acc = fmaf(o[k] * inv, sW3[k * NC + j], acc);
        lg[j] = acc;
        mx = fmaxf(mx, acc);
    }
    float s = 0.0f;
    for (int j = 0; j < NC; ++j) { lg[j] = __expf(lg[j] - mx); s += lg[j]; }
    float is = 1.0f / s;
    float* op = out + (size_t)i * NC;
    for (int j = 0; j < NC; ++j) op[j] = lg[j] * is;
}

// =====================================================================
extern "C" void kernel_launch(void* const* d_in, const int* in_sizes, int n_in,
                              void* d_out, int out_size, void* d_ws, size_t ws_size,
                              hipStream_t stream) {
    const float* x   = (const float*)d_in[0];
    const int*   ei  = (const int*)  d_in[1];   // [2, E] int32
    const float* W1  = (const float*)d_in[2];
    const float* b1  = (const float*)d_in[3];
    const float* Wl  = (const float*)d_in[4];
    const float* bl  = (const float*)d_in[5];
    const float* Wr  = (const float*)d_in[6];
    const float* br  = (const float*)d_in[7];
    const float* W3  = (const float*)d_in[8];
    const float* b3  = (const float*)d_in[9];
    float* out = (float*)d_out;

    char* ws = (char*)d_ws;
    unsigned short* xbf  = (unsigned short*)(ws + O_XBF);
    unsigned short* frag = (unsigned short*)(ws + O_FRAG);
    float* xw  = (float*)(ws + O_XW);
    float* h   = (float*)(ws + O_H);
    float* agg = (float*)(ws + O_AGG);
    float* deg = (float*)(ws + O_DEG);
    float* cnt = (float*)(ws + O_CNT);
    float* dis = (float*)(ws + O_DIS);

    k_conv_x     <<<(NN * KPAD) / 256, 256, 0, stream>>>(x, xbf);
    k_pack_w1    <<<(NCH * 2 * 32 * 16) / 256, 256, 0, stream>>>(W1, frag);
    k_init       <<<(NN * H1) / 256, 256, 0, stream>>>(deg, cnt, agg);
    k_deg        <<<EE / 256, 256, 0, stream>>>(ei, deg);
    k_dis        <<<NN / 256, 256, 0, stream>>>(deg, dis);
    k_gemm       <<<NN / 64, 128, 92160, stream>>>(xbf, frag, xw);
    k_selfh      <<<(NN * H1) / 256, 256, 0, stream>>>(dis, xw, h);
    k_gcn_edges  <<<(EE * 8) / 256, 256, 0, stream>>>(ei, dis, xw, h);
    k_bias_relu  <<<(NN * H1) / 256, 256, 0, stream>>>(h, b1);
    k_sage_edges <<<(EE * 8) / 256, 256, 0, stream>>>(ei, h, agg, cnt);
    k_final      <<<NN / 128, 128, 0, stream>>>(h, agg, cnt, Wl, bl, Wr, br, W3, b3, out);
}
